// DeformConv2dModule_26079041421790
// MI455X (gfx1250) — matrix operand order repaired and checked by reading, not compiled, not run on hardware
//
#include <hip/hip_runtime.h>

// ---------------------------------------------------------------------------
// Deformable Conv2d (B=8, C=256->256, 64x64, k=3, s=1, p=1) + ReLU for gfx1250
//
// Strategy (see analysis): fused bilinear-im2col -> LDS (bf16) -> WMMA bf16
// GEMM with fp32 accumulate. x fits in the 192MB L2, so the scatter/gather
// sampling is L2-bound; the contraction (38.6 GFLOP) runs on
// v_wmma_f32_16x16x32_bf16 (K=32/instr, 8x the f32-WMMA matrix rate).
// ---------------------------------------------------------------------------

typedef __attribute__((ext_vector_type(16))) __bf16 v16bf;
typedef __attribute__((ext_vector_type(8)))  float  v8f;

struct __align__(16) U4 { unsigned int x, y, z, w; };
union Frag { v16bf v; U4 u[2]; };

static __device__ __forceinline__ unsigned short f2bf(float f) {
  // round-to-nearest-even f32 -> bf16 (inputs are finite random normals)
  unsigned int u = __float_as_uint(f);
  u += 0x7FFFu + ((u >> 16) & 1u);
  return (unsigned short)(u >> 16);
}

// ---------------------------------------------------------------------------
// Kernel 1: weight transform  W[co][c][k2] (f32) -> Wt[co][k2*256 + c] (bf16)
// k2-major K ordering means each 32-wide K chunk of the GEMM stays inside one
// tap, so the sampling stage handles exactly one (tap, 32-channel) slice.
// ---------------------------------------------------------------------------
__global__ __launch_bounds__(256) void dcn_wtrans_kernel(
    const float* __restrict__ w, unsigned short* __restrict__ wt) {
  int i = blockIdx.x * 256 + threadIdx.x;        // over 256*2304
  if (i >= 256 * 2304) return;
  int co = i / 2304;
  int t  = i % 2304;
  int k2 = t >> 8;          // t / 256
  int c  = t & 255;         // t % 256
  wt[co * 2304 + t] = f2bf(w[(co * 256 + c) * 9 + k2]);
}

// ---------------------------------------------------------------------------
// Kernel 2: one block per output row (b, h): 64 pixels x 256 couts.
// 8 waves; wave `wv` computes couts [wv*32, wv*32+32) x all 64 pixels
// = 2x4 WMMA tiles (M=cout, N=pixel so stores are row-contiguous).
// ---------------------------------------------------------------------------
__global__ __launch_bounds__(256) void dcn_main_kernel(
    const float* __restrict__ x, const float* __restrict__ offs,
    const unsigned short* __restrict__ wt, float* __restrict__ out) {

  __shared__ int   s_addr[576][4];   // 9 taps x 64 pixels: 4 corner offsets
  __shared__ float s_wgt[576][4];    // 4 corner weights (validity folded in)
  __shared__ __align__(16) unsigned short sW[256 * 32];  // weights  [co][k]
  __shared__ __align__(16) unsigned short sB[64 * 32];   // samples  [pix][k]

  const int tid = threadIdx.x;
  const int pt  = blockIdx.x;       // 0..511
  const int b   = pt >> 6;
  const int h   = pt & 63;

  // ---- bilinear tap precompute: 9 taps x 64 pixels --------------------
  for (int i = tid; i < 576; i += 256) {
    const int k2 = i >> 6;
    const int p  = i & 63;
    const float offy = offs[(((b * 18) + 2 * k2) * 64 + h) * 64 + p];
    const float offx = offs[(((b * 18) + 2 * k2 + 1) * 64 + h) * 64 + p];
    const float py = (float)(h + (k2 / 3) - 1) + offy;
    const float px = (float)(p + (k2 % 3) - 1) + offx;
    const float y0f = floorf(py), x0f = floorf(px);
    const int   y0 = (int)y0f, x0 = (int)x0f;
    const float fy = py - y0f, fx = px - x0f;
    const int   ys[2] = {y0, y0 + 1};
    const int   xs[2] = {x0, x0 + 1};
    const float wy[2] = {1.0f - fy, fy};
    const float wx[2] = {1.0f - fx, fx};
#pragma unroll
    for (int cy = 0; cy < 2; ++cy)
#pragma unroll
      for (int cx = 0; cx < 2; ++cx) {
        const int  yy = ys[cy], xx = xs[cx];
        const bool valid = (yy >= 0) & (yy < 64) & (xx >= 0) & (xx < 64);
        const int  yc = min(max(yy, 0), 63), xc = min(max(xx, 0), 63);
        s_addr[i][cy * 2 + cx] = yc * 64 + xc;
        s_wgt[i][cy * 2 + cx]  = valid ? wy[cy] * wx[cx] : 0.0f;
      }
  }
  __syncthreads();

  const int wave  = tid >> 5;
  const int lane  = tid & 31;
  const int lhalf = lane & 15;
  const int hsel  = lane >> 4;   // 0: lanes 0-15, 1: lanes 16-31

  v8f zero = {};
  v8f acc[8];
#pragma unroll
  for (int i = 0; i < 8; ++i) acc[i] = zero;

  // per-thread sampling assignment: pixel sp, 8 channels of subgroup scg
  const int sp  = tid & 63;
  const int scg = tid >> 6;      // 0..3

  for (int kk = 0; kk < 72; ++kk) {
    const int k2 = kk >> 3;
    const int c0 = (kk & 7) << 5;

    __syncthreads();   // previous iteration's fragment reads done

    // ---- stage weight tile: thread == cout row, 32 bf16 (64B) --------
    {
      const U4* src = (const U4*)(wt + (size_t)tid * 2304 + k2 * 256 + c0);
      U4* dst = (U4*)(sW + tid * 32);
      dst[0] = src[0]; dst[1] = src[1]; dst[2] = src[2]; dst[3] = src[3];
    }

    // ---- stage sampled-value tile (bilinear gather, L2-resident x) ---
    {
      const int   ti  = k2 * 64 + sp;
      const int   a00 = s_addr[ti][0], a01 = s_addr[ti][1];
      const int   a10 = s_addr[ti][2], a11 = s_addr[ti][3];
      const float w00 = s_wgt[ti][0], w01 = s_wgt[ti][1];
      const float w10 = s_wgt[ti][2], w11 = s_wgt[ti][3];
      union { unsigned short s[8]; U4 u; } tmp;
      const float* plane = x + (size_t)(b * 256 + c0 + scg * 8) * 4096;
#pragma unroll
      for (int j = 0; j < 8; ++j) {
        const float v = w00 * plane[a00] + w01 * plane[a01] +
                        w10 * plane[a10] + w11 * plane[a11];
        tmp.s[j] = f2bf(v);
        plane += 4096;
      }
      *(U4*)(sB + sp * 32 + scg * 8) = tmp.u;
    }

    __syncthreads();

    // ---- fragment loads per ISA 16-bit A/B VGPR layouts --------------
    // A (16x32): lane row M=lhalf; halves 0-7 -> K=kbase..kbase+7,
    //            halves 8-15 -> K=kbase+16..kbase+23, kbase = hsel ? 8 : 0
    // B (32x16): lane col N=lhalf; same K mapping as A, koff = hsel*8
    const int kbaseA = hsel << 3;
    const int koffB  = hsel << 3;
    Frag afrag[2], bfrag[4];
#pragma unroll
    for (int ci = 0; ci < 2; ++ci) {
      const unsigned short* pa = sW + (wave * 32 + ci * 16 + lhalf) * 32 + kbaseA;
      afrag[ci].u[0] = *(const U4*)pa;
      afrag[ci].u[1] = *(const U4*)(pa + 16);
    }
#pragma unroll
    for (int pi = 0; pi < 4; ++pi) {
      const unsigned short* pb = sB + (pi * 16 + lhalf) * 32 + koffB;
      bfrag[pi].u[0] = *(const U4*)pb;
      bfrag[pi].u[1] = *(const U4*)(pb + 16);
    }

    // ---- 8 WMMAs, A fragment reused across the 4 pixel tiles ---------
#pragma unroll
    for (int ci = 0; ci < 2; ++ci)
#pragma unroll
      for (int pi = 0; pi < 4; ++pi)
        acc[ci * 4 + pi] = __builtin_amdgcn_wmma_f32_16x16x32_bf16(
            false, afrag[ci].v, false, bfrag[pi].v,
            (short)0, acc[ci * 4 + pi], false, false);
  }

  // ---- epilogue: ReLU + store. D layout: VGPR r holds rows M=r (lanes
  // 0-15) and M=r+8 (lanes 16-31), N = lhalf = pixel -> contiguous stores.
#pragma unroll
  for (int ci = 0; ci < 2; ++ci)
#pragma unroll
    for (int pi = 0; pi < 4; ++pi) {
      const v8f a = acc[ci * 4 + pi];
#pragma unroll
      for (int r = 0; r < 8; ++r) {
        const int co = wave * 32 + ci * 16 + hsel * 8 + r;
        const int p  = pi * 16 + lhalf;
        out[(((size_t)b * 256 + co) * 64 + h) * 64 + p] = fmaxf(a[r], 0.0f);
      }
    }
}

// ---------------------------------------------------------------------------
extern "C" void kernel_launch(void* const* d_in, const int* in_sizes, int n_in,
                              void* d_out, int out_size, void* d_ws, size_t ws_size,
                              hipStream_t stream) {
  const float* x      = (const float*)d_in[0];   // [8,256,64,64]
  const float* offs   = (const float*)d_in[1];   // [8,18,64,64]
  const float* weight = (const float*)d_in[2];   // [256,256,3,3]
  float*       out    = (float*)d_out;           // [8,256,64,64]
  unsigned short* wt  = (unsigned short*)d_ws;   // 256*2304 bf16 = 1.18 MB

  dcn_wtrans_kernel<<<2304, 256, 0, stream>>>(weight, wt);
  dcn_main_kernel<<<512, 256, 0, stream>>>(x, offs, wt, out);
}
